// CrossFocusedLinearAttentionPrune_62079457297002
// MI455X (gfx1250) — matrix operand order repaired and checked by reading, not compiled, not run on hardware
//
#include <hip/hip_runtime.h>

#define B_ 8
#define N_ 4096
#define M_ 4096
#define C_ 256
#define EPS_ 1e-6f

typedef __bf16 bf16;
typedef __attribute__((ext_vector_type(16))) __bf16 v16bf;
typedef __attribute__((ext_vector_type(8)))  float  v8f;
typedef __attribute__((ext_vector_type(4)))  int    v4i;

#if __has_builtin(__builtin_amdgcn_global_load_async_to_lds_b128)
#define HAVE_ASYNC_LDS 1
#else
#define HAVE_ASYNC_LDS 0
#warning "PROBE: no __builtin_amdgcn_global_load_async_to_lds_b128"
#endif

// 16-byte global->LDS copy: async (ASYNCcnt-tracked) on CDNA5
__device__ inline void cp16(bf16* dst_lds, const bf16* src_g) {
#if HAVE_ASYNC_LDS
  __builtin_amdgcn_global_load_async_to_lds_b128(
      (__attribute__((address_space(1))) v4i*)(src_g),
      (__attribute__((address_space(3))) v4i*)(dst_lds), 0, 0);
#else
  *(uint4*)dst_lds = *(const uint4*)src_g;
#endif
}
__device__ inline void async_wait0() {
#if HAVE_ASYNC_LDS
#if __has_builtin(__builtin_amdgcn_s_wait_asynccnt)
  __builtin_amdgcn_s_wait_asynccnt(0);
#else
  asm volatile("s_wait_asynccnt 0" ::: "memory");
#endif
#endif
}

// ---- WMMA fragment helpers (layouts per CDNA5 ISA 7.12.2) ----
__device__ inline v16bf frag_a(const bf16* p, int row0, int ld, int k0) {
  int lane = threadIdx.x & 31;
  int half = lane >> 4, l = lane & 15;
  v16bf f;
#pragma unroll
  for (int i = 0; i < 16; ++i) {
    int k = (i < 8 ? i : i + 8) + half * 8;
    f[i] = p[(row0 + l) * ld + k0 + k];
  }
  return f;
}
__device__ inline v16bf frag_b(const bf16* p, int col0, int ld, int k0) {
  int lane = threadIdx.x & 31;
  int half = lane >> 4, l = lane & 15;
  v16bf f;
#pragma unroll
  for (int i = 0; i < 16; ++i) {
    int k = (i < 8 ? i : i + 8) + half * 8;
    f[i] = p[(col0 + l) * ld + k0 + k];
  }
  return f;
}
__device__ inline v8f wmma_bf16(v16bf a, v16bf b, v8f c) {
  return __builtin_amdgcn_wmma_f32_16x16x32_bf16(false, a, false, b, (short)0, c, false, false);
}

// ---- Kernel 0: pre-convert the 4 weight matrices to bf16 ----
__global__ __launch_bounds__(256) void cvt_w_kernel(const float* __restrict__ Wq, const float* __restrict__ Wk,
                                                    const float* __restrict__ Wv, const float* __restrict__ Wp,
                                                    bf16* __restrict__ wb) {
  int w = blockIdx.y;
  const float* src = (w == 0 ? Wq : w == 1 ? Wk : w == 2 ? Wv : Wp);
  int i = blockIdx.x * 256 + threadIdx.x;
  wb[w * 65536 + i] = (bf16)src[i];
}

// ---- Kernel 1: q/k/v projections + focusing epilogue in registers ----
__global__ __launch_bounds__(256) void qkv_proj_kernel(
    const float* __restrict__ query, const float* __restrict__ key_, const float* __restrict__ value,
    const bf16* __restrict__ wb, const float* __restrict__ scale,
    float* __restrict__ q_f32, bf16* __restrict__ q_bf, bf16* __restrict__ k_bf, bf16* __restrict__ v_bf)
{
  __shared__ __align__(16) bf16 As[32 * 256];
  __shared__ __align__(16) bf16 Bs[2][256 * 32];   // [d][k] double-buffered
  __shared__ float red[32 * 4];

  int tid = threadIdx.x;
  int b = blockIdx.y, which = blockIdx.z;
  int n0 = blockIdx.x * 32;
  const float* in = (which == 0 ? query : which == 1 ? key_ : value) + (size_t)b * N_ * C_;
  const bf16* W = wb + which * 65536;

  for (int i = tid; i < 32 * 256; i += 256)
    As[i] = (bf16)in[(size_t)(n0 + (i >> 8)) * 256 + (i & 255)];

  int wave = tid >> 5, rowt = (wave & 1) * 16, colb = (wave >> 1) * 64;
  v8f acc[4] = {};

  auto stageB = [&](int buf, int kk) {
#pragma unroll
    for (int t = 0; t < 4; ++t) {
      int piece = tid + t * 256;
      int d = piece >> 2, sub = piece & 3;
      cp16(&Bs[buf][d * 32 + sub * 8], W + d * 256 + kk + sub * 8);
    }
  };
  stageB(0, 0);
  for (int it = 0; it < 8; ++it) {
    int cur = it & 1;
    async_wait0();
    __syncthreads();
    if (it < 7) stageB(cur ^ 1, (it + 1) * 32);
    v16bf a = frag_a(As, rowt, 256, it * 32);
#pragma unroll
    for (int ct = 0; ct < 4; ++ct)
      acc[ct] = wmma_bf16(a, frag_b(&Bs[cur][0], colb + ct * 16, 32, 0), acc[ct]);
  }

  int lane = tid & 31, half = lane >> 4, l = lane & 15;
  size_t gbase = (size_t)(b * N_ + n0) * 256;

  if (which < 2) {
    float sp[4];
#pragma unroll
    for (int ct = 0; ct < 4; ++ct) sp[ct] = log1pf(expf(scale[colb + ct * 16 + l]));
    float s2[8];
#pragma unroll
    for (int r = 0; r < 8; ++r) {
      float ss = 0.f;
#pragma unroll
      for (int ct = 0; ct < 4; ++ct) {
        float v = (fmaxf(acc[ct][r], 0.f) + EPS_) / sp[ct];
        acc[ct][r] = v;
        ss += v * v;
      }
      s2[r] = ss;
    }
#pragma unroll
    for (int r = 0; r < 8; ++r) {
#pragma unroll
      for (int mask = 1; mask <= 8; mask <<= 1) s2[r] += __shfl_xor(s2[r], mask, 32);
    }
    if (l == 0) {
#pragma unroll
      for (int r = 0; r < 8; ++r) red[(rowt + half * 8 + r) * 4 + (wave >> 1)] = s2[r];
    }
    __syncthreads();
#pragma unroll
    for (int r = 0; r < 8; ++r) {
      int row = rowt + half * 8 + r;
      float nr = sqrtf(red[row * 4 + 0] + red[row * 4 + 1] + red[row * 4 + 2] + red[row * 4 + 3]);
#pragma unroll
      for (int ct = 0; ct < 4; ++ct) {
        float v = acc[ct][r];
        v = ((v * v * v) / nr) * nr;   // faithful to reference ordering
        size_t gi = gbase + (size_t)row * 256 + colb + ct * 16 + l;
        if (which == 0) { q_f32[gi] = v; q_bf[gi] = (bf16)v; }
        else            k_bf[gi] = (bf16)v;
      }
    }
  } else {
#pragma unroll
    for (int ct = 0; ct < 4; ++ct)
#pragma unroll
      for (int r = 0; r < 8; ++r)
        v_bf[gbase + (size_t)(rowt + half * 8 + r) * 256 + colb + ct * 16 + l] = (bf16)acc[ct][r];
  }
}

// ---- Kernel 2: k_sum[b,c] (deterministic) ----
__global__ __launch_bounds__(256) void ksum_kernel(const bf16* __restrict__ k_bf, float* __restrict__ ksum) {
  int b = blockIdx.x, c = threadIdx.x;
  const bf16* p = k_bf + (size_t)b * M_ * C_ + c;
  float s = 0.f;
  for (int m = 0; m < M_; ++m) s += (float)p[(size_t)m * C_];
  ksum[b * C_ + c] = s;
}

// ---- Kernel 3: kv partials, split-K over m (4 slices of 1024) ----
__global__ __launch_bounds__(256) void kv_kernel(const bf16* __restrict__ k_bf, const bf16* __restrict__ v_bf,
                                                 float* __restrict__ kv_part) {
  __shared__ __align__(16) bf16 Ks[64 * 64];   // [c][m], ld=64
  __shared__ __align__(16) bf16 Vs[256 * 64];  // [d][m], ld=64
  int tid = threadIdx.x, c0 = blockIdx.x * 64, slice = blockIdx.y, b = blockIdx.z;
  int wave = tid >> 5, rowt = (wave & 3) * 16, colb = (wave >> 2) * 128;
  v8f acc[8] = {};
  const bf16* kb = k_bf + (size_t)b * M_ * C_;
  const bf16* vb = v_bf + (size_t)b * M_ * C_;
  int mbeg = slice * 1024;
  for (int m0 = mbeg; m0 < mbeg + 1024; m0 += 64) {
    __syncthreads();
    for (int i = tid; i < 64 * 64; i += 256) {
      int m = i >> 6, c = i & 63;
      Ks[c * 64 + m] = kb[(size_t)(m0 + m) * 256 + c0 + c];
    }
    for (int i = tid; i < 256 * 64; i += 256) {
      int m = i >> 8, d = i & 255;
      Vs[d * 64 + m] = vb[(size_t)(m0 + m) * 256 + d];
    }
    __syncthreads();
#pragma unroll
    for (int k0 = 0; k0 < 64; k0 += 32) {
      v16bf a = frag_a(Ks, rowt, 64, k0);
#pragma unroll
      for (int ct = 0; ct < 8; ++ct)
        acc[ct] = wmma_bf16(a, frag_b(Vs, colb + ct * 16, 64, k0), acc[ct]);
    }
  }
  int lane = tid & 31, half = lane >> 4, l = lane & 15;
  float* outp = kv_part + ((size_t)slice * 8 + b) * 65536;
#pragma unroll
  for (int ct = 0; ct < 8; ++ct)
#pragma unroll
    for (int r = 0; r < 8; ++r)
      outp[(size_t)(c0 + rowt + half * 8 + r) * 256 + colb + ct * 16 + l] = acc[ct][r];
}

// ---- Kernel 4: reduce split-K partials -> kv^T in bf16 ----
__global__ __launch_bounds__(256) void kv_red_kernel(const float* __restrict__ kv_part, bf16* __restrict__ kvT_bf) {
  int b = blockIdx.y, c = blockIdx.x, d = threadIdx.x;
  size_t o = (size_t)c * 256 + d;
  float s = 0.f;
#pragma unroll
  for (int sl = 0; sl < 4; ++sl) s += kv_part[((size_t)sl * 8 + b) * 65536 + o];
  kvT_bf[(size_t)b * 65536 + (size_t)d * 256 + c] = (bf16)s;
}

// ---- Kernel 5: x = (q @ kv) * z ----
__global__ __launch_bounds__(256) void xqkv_kernel(const bf16* __restrict__ q_bf, const float* __restrict__ q_f32,
                                                   const bf16* __restrict__ kvT_bf, const float* __restrict__ ksum,
                                                   float* __restrict__ xbuf) {
  __shared__ __align__(16) bf16 As[32 * 256];
  __shared__ __align__(16) bf16 Bs[2][256 * 32];  // kv^T chunk [d][k]
  __shared__ float red2[256];
  __shared__ float zrow[32];
  int tid = threadIdx.x, b = blockIdx.y, n0 = blockIdx.x * 32;
  int wave = tid >> 5, rowt = (wave & 1) * 16, colb = (wave >> 1) * 64;
  size_t gbase = (size_t)(b * N_ + n0) * 256;
  const bf16* KVT = kvT_bf + (size_t)b * 65536;
  for (int i = tid; i < 32 * 256; i += 256) As[i] = q_bf[gbase + i];
  v8f acc[4] = {};
  auto stageB = [&](int buf, int kk) {
#pragma unroll
    for (int t = 0; t < 4; ++t) {
      int piece = tid + t * 256;
      int d = piece >> 2, sub = piece & 3;
      cp16(&Bs[buf][d * 32 + sub * 8], KVT + d * 256 + kk + sub * 8);
    }
  };
  stageB(0, 0);
  for (int it = 0; it < 8; ++it) {
    int cur = it & 1;
    async_wait0();
    __syncthreads();
    if (it < 7) stageB(cur ^ 1, (it + 1) * 32);
    v16bf a = frag_a(As, rowt, 256, it * 32);
#pragma unroll
    for (int ct = 0; ct < 4; ++ct)
      acc[ct] = wmma_bf16(a, frag_b(&Bs[cur][0], colb + ct * 16, 32, 0), acc[ct]);
  }
  __syncthreads();
  int row = tid >> 3, cs = (tid & 7) * 32;
  float zp = 0.f;
  for (int j = 0; j < 32; ++j) {
    int c = cs + j;
    zp += q_f32[gbase + (size_t)row * 256 + c] * ksum[b * 256 + c];
  }
  red2[tid] = zp;
  __syncthreads();
  if (tid < 32) {
    float s = 0.f;
#pragma unroll
    for (int j = 0; j < 8; ++j) s += red2[tid * 8 + j];
    zrow[tid] = 1.f / (s + EPS_);
  }
  __syncthreads();
  int lane = tid & 31, half = lane >> 4, l = lane & 15;
#pragma unroll
  for (int ct = 0; ct < 4; ++ct)
#pragma unroll
    for (int r = 0; r < 8; ++r) {
      int rr = rowt + half * 8 + r;
      xbuf[gbase + (size_t)rr * 256 + colb + ct * 16 + l] = acc[ct][r] * zrow[rr];
    }
}

// ---- Kernel 6: depthwise 5x5 conv + bias + residual(q) -> bf16 y ----
__global__ __launch_bounds__(256) void dwconv_kernel(const float* __restrict__ xbuf, const float* __restrict__ q_f32,
                                                     const float* __restrict__ dwc_w, const float* __restrict__ dwc_b,
                                                     bf16* __restrict__ ybuf) {
  __shared__ float tile[68 * 68];
  int c = blockIdx.x, b = blockIdx.y, tid = threadIdx.x;
  const float* xp = xbuf + (size_t)b * N_ * C_ + c;
  for (int i = tid; i < 68 * 68; i += 256) {
    int h = i / 68 - 2, w = i % 68 - 2;
    float v = 0.f;
    if (h >= 0 && h < 64 && w >= 0 && w < 64) v = xp[(size_t)(h * 64 + w) * 256];
    tile[i] = v;
  }
  float wgt[25];
#pragma unroll
  for (int j = 0; j < 25; ++j) wgt[j] = dwc_w[c * 25 + j];
  float bias = dwc_b[c];
  __syncthreads();
  for (int p = tid; p < 4096; p += 256) {
    int h = p >> 6, w = p & 63;
    float a = bias;
#pragma unroll
    for (int kh = 0; kh < 5; ++kh)
#pragma unroll
      for (int kw = 0; kw < 5; ++kw)
        a += tile[(h + kh) * 68 + (w + kw)] * wgt[kh * 5 + kw];
    a += q_f32[((size_t)b * N_ + p) * 256 + c];
    ybuf[((size_t)b * N_ + p) * 256 + c] = (bf16)a;
  }
}

// ---- Kernel 7: out = y @ Wproj^T + bproj ----
__global__ __launch_bounds__(256) void proj_out_kernel(const bf16* __restrict__ ybuf, const bf16* __restrict__ wb,
                                                       const float* __restrict__ bproj, float* __restrict__ out) {
  __shared__ __align__(16) bf16 As[32 * 256];
  __shared__ __align__(16) bf16 Bs[2][256 * 32];
  int tid = threadIdx.x, b = blockIdx.y, n0 = blockIdx.x * 32;
  int wave = tid >> 5, rowt = (wave & 1) * 16, colb = (wave >> 1) * 64;
  size_t gbase = (size_t)(b * N_ + n0) * 256;
  const bf16* W = wb + 3 * 65536;
  for (int i = tid; i < 32 * 256; i += 256) As[i] = ybuf[gbase + i];
  v8f acc[4] = {};
  auto stageB = [&](int buf, int kk) {
#pragma unroll
    for (int t = 0; t < 4; ++t) {
      int piece = tid + t * 256;
      int d = piece >> 2, sub = piece & 3;
      cp16(&Bs[buf][d * 32 + sub * 8], W + d * 256 + kk + sub * 8);
    }
  };
  stageB(0, 0);
  for (int it = 0; it < 8; ++it) {
    int cur = it & 1;
    async_wait0();
    __syncthreads();
    if (it < 7) stageB(cur ^ 1, (it + 1) * 32);
    v16bf a = frag_a(As, rowt, 256, it * 32);
#pragma unroll
    for (int ct = 0; ct < 4; ++ct)
      acc[ct] = wmma_bf16(a, frag_b(&Bs[cur][0], colb + ct * 16, 32, 0), acc[ct]);
  }
  int lane = tid & 31, half = lane >> 4, l = lane & 15;
#pragma unroll
  for (int ct = 0; ct < 4; ++ct)
#pragma unroll
    for (int r = 0; r < 8; ++r) {
      int col = colb + ct * 16 + l;
      out[gbase + (size_t)(rowt + half * 8 + r) * 256 + col] = acc[ct][r] + bproj[col];
    }
}

extern "C" void kernel_launch(void* const* d_in, const int* in_sizes, int n_in,
                              void* d_out, int out_size, void* d_ws, size_t ws_size,
                              hipStream_t stream) {
  const float* query = (const float*)d_in[0];
  const float* key_  = (const float*)d_in[1];
  const float* value = (const float*)d_in[2];
  const float* Wq    = (const float*)d_in[3];
  const float* Wk    = (const float*)d_in[4];
  const float* Wv    = (const float*)d_in[5];
  const float* Wproj = (const float*)d_in[6];
  const float* bproj = (const float*)d_in[7];
  const float* dwc_w = (const float*)d_in[8];
  const float* dwc_b = (const float*)d_in[9];
  const float* scale = (const float*)d_in[10];
  float* out = (float*)d_out;

  char* ws = (char*)d_ws;
  float* q_f32  = (float*)(ws);                 // 33.5 MB
  float* xbuf   = (float*)(ws + 33554432);      // 33.5 MB
  bf16*  q_bf   = (bf16*)(ws + 67108864);       // 16.8 MB
  bf16*  k_bf   = (bf16*)(ws + 83886080);       // 16.8 MB
  bf16*  v_bf   = (bf16*)(ws + 100663296);      // 16.8 MB
  bf16*  ybuf   = (bf16*)(ws + 117440512);      // 16.8 MB
  bf16*  kvT_bf = (bf16*)(ws + 134217728);      // 1 MB
  float* kv_part= (float*)(ws + 135266304);     // 8 MB
  float* ksum   = (float*)(ws + 143654912);     // 8 KB
  bf16*  wbf    = (bf16*)(ws + 143663104);      // 512 KB

  cvt_w_kernel<<<dim3(256, 4), 256, 0, stream>>>(Wq, Wk, Wv, Wproj, wbf);
  qkv_proj_kernel<<<dim3(128, 8, 3), 256, 0, stream>>>(query, key_, value, wbf, scale,
                                                       q_f32, q_bf, k_bf, v_bf);
  ksum_kernel<<<dim3(8), 256, 0, stream>>>(k_bf, ksum);
  kv_kernel<<<dim3(4, 4, 8), 256, 0, stream>>>(k_bf, v_bf, kv_part);
  kv_red_kernel<<<dim3(256, 8), 256, 0, stream>>>(kv_part, kvT_bf);
  xqkv_kernel<<<dim3(128, 8), 256, 0, stream>>>(q_bf, q_f32, kvT_bf, ksum, xbuf);
  dwconv_kernel<<<dim3(256, 8), 256, 0, stream>>>(xbuf, q_f32, dwc_w, dwc_b, ybuf);
  proj_out_kernel<<<dim3(128, 8), 256, 0, stream>>>(ybuf, wbf, bproj, out);
}
